// ASA_12275016532647
// MI455X (gfx1250) — matrix operand order repaired and checked by reading, not compiled, not run on hardware
//
#include <hip/hip_runtime.h>
#include <hip/hip_bf16.h>
#include <math.h>

// ---------------------------------------------------------------------------
// ASA block (axial self-attention) for gfx1250 / CDNA5, wave32 + WMMA + TDM.
//   k1: x1 = BNPReLU(w1 @ x) -> f16 [b,t,f,c4];  x2 = BNPReLU(w2 @ x) -> f16 [b,f,t,c4]
//   k2: freq attention  (per b,t):  of = softmax(X X^T / 128) X        -> f16 [b,f,c4,t]
//   k3: time attention  (per b,f):  m  = causal_softmax(Y Y^T / s) V   -> f16 [b,f,t,c4]
//   k4: out = BNPReLU(w3 @ m) + x                                      -> f32 [b,c,t,f]
// All matmuls use v_wmma_f32_16x16x32_f16 (K=32 native). k2 stages its X slab
// into LDS with the Tensor Data Mover (tensor_load_to_lds + s_wait_tensorcnt).
// ---------------------------------------------------------------------------

#define DEVINL __device__ __forceinline__

typedef _Float16 h16;
typedef __attribute__((ext_vector_type(16))) _Float16 v16h;
typedef __attribute__((ext_vector_type(8)))  float    v8f;
typedef __attribute__((ext_vector_type(4)))  unsigned int u32x4;
typedef __attribute__((ext_vector_type(4)))  int          i32x4;
typedef __attribute__((ext_vector_type(8)))  int          i32x8;

namespace asa {
constexpr int B  = 2;
constexpr int C  = 128;
constexpr int C4 = 32;
constexpr int T  = 512;
constexpr int F  = 256;
constexpr int TF = T * F;
constexpr float EPS  = 1e-5f;
constexpr float MASK = -1000000000.0f;
constexpr float SC1  = 1.0f / 128.0f;              // 1/sqrt(C*F/2)
constexpr float SC2  = 0.0055242717280199026f;     // 1/sqrt(C*T/2)
constexpr size_t NB  = (size_t)B * T * F * C4;     // elements per f16 scratch tensor
}

DEVINL v8f wmma16(v16h a, v16h b, v8f c) {
  // D = A(16x32 f16) * B(32x16 f16) + C(16x16 f32)
  return __builtin_amdgcn_wmma_f32_16x16x32_f16(false, a, false, b,
                                                (short)0, c, false, false);
}

// A-operand (16x32 f16) lane packing: lane L holds row M=L&15.
//   elements 0..7  -> K = 2*koffA + 0..7     (koffA = (L&16)?8:0)
//   elements 8..15 -> K = 2*koffA + 8 + 0..7
DEVINL v16h pack_a(const h16* row, int koffA) {
  v16h a;
#pragma unroll
  for (int e = 0; e < 8; ++e) { a[e] = row[2 * koffA + e]; a[8 + e] = row[2 * koffA + 8 + e]; }
  return a;
}

// ---------------------------------------------------------------------------
// TDM: DMA a contiguous run of nelem 16-bit elements from global to LDS.
// D# built per CDNA5 ISA (1 x nelem 2D tile).
// ---------------------------------------------------------------------------
DEVINL void tdm_load_row_u16(unsigned lds_byte_addr, const void* gptr, int nelem) {
  unsigned long long ga = (unsigned long long)gptr;
  u32x4 g0;
  g0[0] = 1u;                                                 // count=1 (user D#)
  g0[1] = lds_byte_addr;                                      // lds_addr (bits 63:32)
  g0[2] = (unsigned)(ga & 0xFFFFFFFFu);                       // global_addr lo
  g0[3] = (unsigned)((ga >> 32) & 0x01FFFFFFu) | 0x80000000u; // addr hi | type=2
  i32x8 g1;
  g1[0] = 1 << 16;                         // data_size=1 -> 2 bytes/elem
  g1[1] = (nelem & 0xFFFF) << 16;          // tensor_dim0[15:0] @ bits 63:48
  g1[2] = (nelem >> 16) | (1 << 16);       // tensor_dim0[31:16] | tensor_dim1=1
  g1[3] = (nelem & 0xFFFF) << 16;          // tile_dim0 @ bits 127:112
  g1[4] = 1;                               // tile_dim1=1
  g1[5] = nelem;                           // tensor_dim0_stride lo
  g1[6] = 0;
  g1[7] = 0;
  i32x4 z4 = {0, 0, 0, 0};
#if defined(__clang_major__) && (__clang_major__ >= 23)
  i32x8 z8 = {0, 0, 0, 0, 0, 0, 0, 0};
  __builtin_amdgcn_tensor_load_to_lds(g0, g1, z4, z4, z8, 0);
#else
  __builtin_amdgcn_tensor_load_to_lds(g0, g1, z4, z4, 0);
#endif
}

// ---------------------------------------------------------------------------
// Kernel 1: pointwise convs 1 & 2 + BN + PReLU.  One wave per 16-position tile.
// ---------------------------------------------------------------------------
__global__ void __launch_bounds__(32)
k1_pw12(const float* __restrict__ x,
        const float* __restrict__ w1, const float* __restrict__ w2,
        const float* __restrict__ g1, const float* __restrict__ b1,
        const float* __restrict__ m1, const float* __restrict__ v1,
        const float* __restrict__ a1,
        const float* __restrict__ g2, const float* __restrict__ b2,
        const float* __restrict__ m2, const float* __restrict__ v2,
        const float* __restrict__ a2,
        h16* __restrict__ x1h, h16* __restrict__ x2h)
{
  using namespace asa;
  __shared__ h16 wl[2][C4][C];                       // 16 KB: w1,w2 in f16
  const int lane = threadIdx.x;
  for (int i = lane; i < 2 * C4 * C; i += 32) {
    float wv = (i < C4 * C) ? w1[i] : w2[i - C4 * C];
    ((h16*)wl)[i] = (h16)wv;
  }
  __syncthreads();

  const int blk = blockIdx.x;                        // B * TF/16 blocks
  const int b   = blk >> 13;
  const int p0  = (blk & 8191) * 16;
  const int t   = p0 >> 8;                           // p0 / F
  const int f0  = p0 & (F - 1);
  const int n     = lane & 15;
  const int koffA = (lane & 16) ? 8 : 0;
  const int koffB = (lane & 16) ? 16 : 0;
  const int mb    = (lane & 16) ? 8 : 0;

  v8f acc[4] = {v8f{}, v8f{}, v8f{}, v8f{}};         // [w1|w2][otile 0|1]
  const float* xb = x + (size_t)b * C * TF + (size_t)t * F + f0 + n;

#pragma unroll
  for (int ks = 0; ks < 4; ++ks) {                   // K = C = 4 * 32
    if (ks < 3) __builtin_prefetch(xb + (size_t)(ks + 1) * 32 * TF, 0, 1);
    v16h bt;                                         // B[k=c, n=pos]: lane = column
    const int cb = ks * 32 + koffB;
#pragma unroll
    for (int e = 0; e < 16; ++e) bt[e] = (h16)xb[(size_t)(cb + e) * TF];
#pragma unroll
    for (int ws = 0; ws < 2; ++ws)
#pragma unroll
      for (int ot = 0; ot < 2; ++ot) {
        v16h at = pack_a(&wl[ws][ot * 16 + n][ks * 32], koffA);
        acc[ws * 2 + ot] = wmma16(at, bt, acc[ws * 2 + ot]);
      }
  }

  const float al1 = a1[0], al2 = a2[0];
#pragma unroll
  for (int j = 0; j < 8; ++j) {
    const int M = mb + j;
#pragma unroll
    for (int ws = 0; ws < 2; ++ws)
#pragma unroll
      for (int ot = 0; ot < 2; ++ot) {
        const int o = ot * 16 + M;
        const float g  = ws ? g2[o] : g1[o];
        const float bb = ws ? b2[o] : b1[o];
        const float mm = ws ? m2[o] : m1[o];
        const float vv = ws ? v2[o] : v1[o];
        float val = (acc[ws * 2 + ot][j] - mm) * (g * rsqrtf(vv + EPS)) + bb;
        val = (val >= 0.f) ? val : (ws ? al2 : al1) * val;
        const h16 hv = (h16)val;
        if (ws == 0)
          x1h[(((size_t)b * T + t) * F + f0 + n) * C4 + o] = hv;     // [b,t,f,c]
        else
          x2h[(((size_t)b * F + (f0 + n)) * T + t) * C4 + o] = hv;   // [b,f,t,c]
      }
  }
}

// ---------------------------------------------------------------------------
// Kernel 2: frequency attention (non-causal). One wave per (b,t,16-f-rows).
// Output of stored TRANSPOSED per (b,f): [b,f,c4,t] so k3 B-tiles are contiguous.
// ---------------------------------------------------------------------------
__global__ void __launch_bounds__(32)
k2_freq(const h16* __restrict__ x1h, h16* __restrict__ ofT)
{
  using namespace asa;
  __shared__ h16   X[F][C4];                          // 16 KB   X[f][c]
  __shared__ float S[16][F];                          // 16 KB   score rows
  __shared__ h16   P[16][F];                          // 8 KB    softmax rows (f16)
  const int lane = threadIdx.x;
  const int blk  = blockIdx.x;                        // B*T*(F/16)
  const int fb   = blk & 15;
  const int t    = (blk >> 4) & (T - 1);
  const int b    = blk >> 13;
  const int m0   = fb * 16;
  const int n     = lane & 15;
  const int hh    = lane >> 4;
  const int koffA = (lane & 16) ? 8 : 0;
  const int koffB = (lane & 16) ? 16 : 0;
  const int mb    = (lane & 16) ? 8 : 0;

  // --- Tensor Data Mover: pull the whole X slab (16 KB contiguous) into LDS.
  {
    const h16* src = x1h + (((size_t)b * T + t) * F) * C4;
    const unsigned lds_addr = (unsigned)(size_t)(&X[0][0]);  // LDS aperture low bits
    tdm_load_row_u16(lds_addr, src, F * C4);
    __builtin_amdgcn_s_wait_tensorcnt(0);
  }
  __syncthreads();

  // S = (X X^T) * SC1 for 16 rows
  v16h aq = pack_a(&X[m0 + n][0], koffA);             // rows constant across n0
#pragma unroll 4
  for (int n0 = 0; n0 < F; n0 += 16) {
    v16h bt;                                          // B[k=c, n=g] = X[n0+n][c]
    const h16* br = &X[n0 + n][koffB];
#pragma unroll
    for (int e = 0; e < 16; ++e) bt[e] = br[e];
    v8f d = {};
    d = wmma16(aq, bt, d);
#pragma unroll
    for (int j = 0; j < 8; ++j) S[mb + j][n0 + n] = d[j] * SC1;
  }
  __syncthreads();

  { // row softmax: lane pairs (n, n+16) split each row, combine via shfl_xor
    const int lo = hh * (F / 2), hi = lo + F / 2;
    float mx = -3.4e38f;
    for (int s = lo; s < hi; ++s) mx = fmaxf(mx, S[n][s]);
    mx = fmaxf(mx, __shfl_xor(mx, 16, 32));
    float sum = 0.f;
    for (int s = lo; s < hi; ++s) { float e = __expf(S[n][s] - mx); sum += e; S[n][s] = e; }
    sum += __shfl_xor(sum, 16, 32);
    const float inv = 1.0f / sum;
    for (int s = lo; s < hi; ++s) P[n][s] = (h16)(S[n][s] * inv);
  }
  __syncthreads();

  // O = P @ X  (16 x 32), K = 256
#pragma unroll
  for (int ct = 0; ct < 2; ++ct) {
    v8f acc = {};
#pragma unroll
    for (int ks = 0; ks < F / 32; ++ks) {
      v16h at = pack_a(&P[n][ks * 32], koffA);
      v16h bt;
      const int gb = ks * 32 + koffB;
#pragma unroll
      for (int e = 0; e < 16; ++e) bt[e] = X[gb + e][ct * 16 + n];
      acc = wmma16(at, bt, acc);
    }
#pragma unroll
    for (int j = 0; j < 8; ++j) {
      const int fr = m0 + mb + j;
      const int c  = ct * 16 + n;
      ofT[(((size_t)b * F + fr) * C4 + c) * T + t] = (h16)acc[j];    // [b,f,c,t]
    }
  }
}

// ---------------------------------------------------------------------------
// Kernel 3: causal time attention + combine. One wave per (b,f,16-t-rows).
// ---------------------------------------------------------------------------
__global__ void __launch_bounds__(32)
k3_time(const h16* __restrict__ x2h, const h16* __restrict__ ofT,
        h16* __restrict__ mh)
{
  using namespace asa;
  __shared__ float S[16][T];                          // 32 KB
  __shared__ h16   P[16][T];                          // 16 KB
  const int lane = threadIdx.x;
  const int blk  = blockIdx.x;                        // B*F*(T/16)
  const int tb   = blk & 31;
  const int f    = (blk >> 5) & (F - 1);
  const int b    = blk >> 13;
  const int t0   = tb * 16;
  const int n     = lane & 15;
  const int hh    = lane >> 4;
  const int koffA = (lane & 16) ? 8 : 0;
  const int koffB = (lane & 16) ? 16 : 0;
  const int mb    = (lane & 16) ? 8 : 0;

  const h16* Xr = x2h + ((size_t)b * F + f) * T * C4;   // [T][C4]
  const h16* Vt = ofT + ((size_t)b * F + f) * C4 * T;   // [C4][T] (transposed V)
  __builtin_prefetch(Vt, 0, 1);

  // S rows: s tiles 0..t0 (causal), mask inside diagonal tile
  v16h aq = pack_a(Xr + (size_t)(t0 + n) * C4, koffA);
  const int ntiles = tb + 1;
  for (int it = 0; it < ntiles; ++it) {
    const int s0 = it * 16;
    v16h bt;
    const h16* br = Xr + (size_t)(s0 + n) * C4 + koffB;
#pragma unroll
    for (int e = 0; e < 16; ++e) bt[e] = br[e];
    v8f d = {};
    d = wmma16(aq, bt, d);
#pragma unroll
    for (int j = 0; j < 8; ++j) {
      const int r = mb + j;
      const int s = s0 + n;
      float val = d[j] * SC2;
      if (s > t0 + r) val = MASK;                     // causal mask
      S[r][s] = val;
    }
  }
  __syncthreads();

  const int Klen = t0 + 16;
  const int Kp   = (Klen + 31) & ~31;                 // pad K to multiple of 32
  { // row softmax over s in [0,Klen): lane pairs split each row
    const int halfl = Klen >> 1;
    const int lo = hh * halfl, hi = lo + halfl;
    float mx = -3.4e38f;
    for (int s = lo; s < hi; ++s) mx = fmaxf(mx, S[n][s]);
    mx = fmaxf(mx, __shfl_xor(mx, 16, 32));
    float sum = 0.f;
    for (int s = lo; s < hi; ++s) { float e = __expf(S[n][s] - mx); sum += e; S[n][s] = e; }
    sum += __shfl_xor(sum, 16, 32);
    const float inv = 1.0f / sum;
    for (int s = lo; s < hi; ++s) P[n][s] = (h16)(S[n][s] * inv);
    if (hh) for (int s = Klen; s < Kp; ++s) P[n][s] = (h16)0.f;  // zero pad
  }
  __syncthreads();

  // M = P @ V  (16 x 32); B-tiles contiguous per lane thanks to [c][t] layout
#pragma unroll
  for (int ct = 0; ct < 2; ++ct) {
    v8f acc = {};
    for (int ks = 0; ks < Kp / 32; ++ks) {
      v16h at = pack_a(&P[n][ks * 32], koffA);
      v16h bt;
      const h16* br = Vt + (size_t)(ct * 16 + n) * T + ks * 32 + koffB;
#pragma unroll
      for (int e = 0; e < 16; ++e) bt[e] = br[e];
      acc = wmma16(at, bt, acc);
    }
#pragma unroll
    for (int j = 0; j < 8; ++j)
      mh[(((size_t)b * F + f) * T + t0 + mb + j) * C4 + ct * 16 + n] = (h16)acc[j];
  }
}

// ---------------------------------------------------------------------------
// Kernel 4: pointwise conv 3 + BN + PReLU + residual. K = 32 -> single WMMA step.
// ---------------------------------------------------------------------------
__global__ void __launch_bounds__(32)
k4_out(const h16* __restrict__ mh, const float* __restrict__ w3,
       const float* __restrict__ g3, const float* __restrict__ b3,
       const float* __restrict__ m3, const float* __restrict__ v3,
       const float* __restrict__ a3,
       const float* __restrict__ x, float* __restrict__ out)
{
  using namespace asa;
  __shared__ h16 W[C][C4];                            // 8 KB  w3 in f16
  const int lane = threadIdx.x;
  for (int i = lane; i < C * C4; i += 32) ((h16*)W)[i] = (h16)w3[i];
  __syncthreads();

  const int blk = blockIdx.x;                         // B*T*(F/16)
  const int fb  = blk & 15;
  const int t   = (blk >> 4) & (T - 1);
  const int b   = blk >> 13;
  const int f0  = fb * 16;
  const int n     = lane & 15;
  const int koffA = (lane & 16) ? 8 : 0;
  const int koffB = (lane & 16) ? 16 : 0;
  const int mb    = (lane & 16) ? 8 : 0;

  v16h bt;                                            // B[k=c, n=f] = m[b,f0+n,t,c]
  const h16* mr = mh + (((size_t)b * F + f0 + n) * T + t) * C4 + koffB;
#pragma unroll
  for (int e = 0; e < 16; ++e) bt[e] = mr[e];

  const float alpha = a3[0];
#pragma unroll
  for (int ot = 0; ot < 8; ++ot) {                    // C = 128 output channels
    v16h at = pack_a(&W[ot * 16 + n][0], koffA);
    v8f d = {};
    d = wmma16(at, bt, d);
#pragma unroll
    for (int j = 0; j < 8; ++j) {
      const int o = ot * 16 + mb + j;
      float val = (d[j] - m3[o]) * (g3[o] * rsqrtf(v3[o] + EPS)) + b3[o];
      val = (val >= 0.f) ? val : alpha * val;
      const size_t idx = (((size_t)b * C + o) * T + t) * F + f0 + n;
      out[idx] = val + x[idx];
    }
  }
}

// ---------------------------------------------------------------------------
extern "C" void kernel_launch(void* const* d_in, const int* in_sizes, int n_in,
                              void* d_out, int out_size, void* d_ws, size_t ws_size,
                              hipStream_t stream) {
  using namespace asa;
  const float* x  = (const float*)d_in[0];
  const float* w1 = (const float*)d_in[1];
  const float* g1 = (const float*)d_in[2];
  const float* b1 = (const float*)d_in[3];
  const float* m1 = (const float*)d_in[4];
  const float* v1 = (const float*)d_in[5];
  const float* a1 = (const float*)d_in[6];
  const float* w2 = (const float*)d_in[7];
  const float* g2 = (const float*)d_in[8];
  const float* b2 = (const float*)d_in[9];
  const float* m2 = (const float*)d_in[10];
  const float* v2 = (const float*)d_in[11];
  const float* a2 = (const float*)d_in[12];
  const float* w3 = (const float*)d_in[13];
  const float* g3 = (const float*)d_in[14];
  const float* b3 = (const float*)d_in[15];
  const float* m3 = (const float*)d_in[16];
  const float* v3 = (const float*)d_in[17];
  const float* a3 = (const float*)d_in[18];
  float* out = (float*)d_out;

  h16* x1h = (h16*)d_ws;          // [b,t,f,c4]  16 MB
  h16* x2h = x1h + NB;            // [b,f,t,c4]  16 MB
  h16* ofT = x2h + NB;            // [b,f,c4,t]  16 MB (transposed values)
  h16* mh  = ofT + NB;            // [b,f,t,c4]  16 MB

  const int blocks = B * TF / 16;                    // 16384
  k1_pw12<<<blocks, 32, 0, stream>>>(x, w1, w2, g1, b1, m1, v1, a1,
                                     g2, b2, m2, v2, a2, x1h, x2h);
  k2_freq<<<B * T * (F / 16), 32, 0, stream>>>(x1h, ofT);
  k3_time<<<B * F * (T / 16), 32, 0, stream>>>(x2h, ofT, mh);
  k4_out<<<B * T * (F / 16), 32, 0, stream>>>(mh, w3, g3, b3, m3, v3, a3, x, out);
}
